// MicroGNN_54004918780322
// MI455X (gfx1250) — matrix operand order repaired and checked by reading, not compiled, not run on hardware
//
#include <hip/hip_runtime.h>
#include <hip/hip_bf16.h>
#include <math.h>

// ---------------- constants ----------------
static const int NN = 50000;   // nodes
static const int EE = 600000;  // edges
// D=256, heads H=2, C=128

typedef __attribute__((ext_vector_type(16))) __bf16 v16bf;
typedef __attribute__((ext_vector_type(8)))  float  v8f;

// ---------------- helpers ----------------
__device__ __forceinline__ unsigned short f2bf(float f) {
    unsigned u = __float_as_uint(f);
    unsigned r = (u + 0x7FFFu + ((u >> 16) & 1u)) >> 16;
    return (unsigned short)r;
}
__device__ __forceinline__ unsigned encf(float f) {
    unsigned u = __float_as_uint(f);
    return (u & 0x80000000u) ? ~u : (u | 0x80000000u);
}
__device__ __forceinline__ float decf(unsigned u) {
    unsigned v = (u & 0x80000000u) ? (u ^ 0x80000000u) : ~u;
    return __uint_as_float(v);
}
__device__ __forceinline__ float gelu_exact(float x) {
    return 0.5f * x * (1.0f + erff(x * 0.70710678118654752f));
}

// ============================================================
// Phase 0: base projection ([N,12]@[12,256]) + sinusoidal PE
// ============================================================
__global__ __launch_bounds__(256) void k_base_proj(
    const float* __restrict__ mx, const float* __restrict__ ms,
    const float* __restrict__ bw, const float* __restrict__ bb,
    const float* __restrict__ pos_scale,
    float* __restrict__ xA, unsigned short* __restrict__ xbf)
{
    int n = blockIdx.x, d = threadIdx.x;
    float acc = bb[d];
    #pragma unroll
    for (int j = 0; j < 5; ++j) acc += mx[n * 5 + j] * bw[j * 256 + d];
    #pragma unroll
    for (int j = 0; j < 7; ++j) acc += ms[n * 7 + j] * bw[(5 + j) * 256 + d];
    const float NEG = -9.210340371976184f / 256.0f; // -ln(10000)/256
    float dv = expf((float)(d & ~1) * NEG);
    float arg = (float)n * dv;
    float pe = (d & 1) ? cosf(arg) : sinf(arg);
    float v = acc + pe * pos_scale[0];
    size_t i = (size_t)n * 256 + d;
    xA[i] = v;
    xbf[i] = f2bf(v);
}

// ============================================================
// Weight packers (f32 -> bf16, transposed to [Nout][K])
// ============================================================
__global__ __launch_bounds__(256) void k_pack_conv(
    const float* __restrict__ cw, const float* __restrict__ cb,
    unsigned short* __restrict__ Wt, float* __restrict__ bias)
{
    int idx = blockIdx.x * 256 + threadIdx.x;       // over 256*768
    if (idx < 256 * 768) {
        int o = idx / 768, rk = idx % 768;
        int kk = rk >> 8, in = rk & 255;            // virtual K = kk*256+in
        Wt[(size_t)o * 768 + rk] = f2bf(cw[((size_t)o * 256 + in) * 3 + kk]);
    }
    if (idx < 256) bias[idx] = cb[idx];
}

__global__ __launch_bounds__(256) void k_pack_qkvs(
    const float* __restrict__ qw, const float* __restrict__ qb,
    const float* __restrict__ kw, const float* __restrict__ kb,
    const float* __restrict__ vw, const float* __restrict__ vb,
    const float* __restrict__ sw, const float* __restrict__ sb,
    unsigned short* __restrict__ Wt, float* __restrict__ bias)
{
    int idx = blockIdx.x * 256 + threadIdx.x;       // over 1024*256
    if (idx < 1024 * 256) {
        int c = idx >> 8, r = idx & 255;
        int sel = c >> 8, cc = c & 255;
        const float* w = (sel == 0) ? qw : (sel == 1) ? kw : (sel == 2) ? vw : sw;
        Wt[(size_t)c * 256 + r] = f2bf(w[(size_t)r * 256 + cc]);
    }
    if (idx < 1024) {
        int sel = idx >> 8, cc = idx & 255;
        const float* b = (sel == 0) ? qb : (sel == 1) ? kb : (sel == 2) ? vb : sb;
        bias[idx] = b[cc];
    }
}

// ============================================================
// Tiled bf16 WMMA GEMM: C[M,Nout] = A[M,K] * Wt^T + bias
//   Wt stored [Nout][K] (pre-transposed), bf16.
//   CONV=1: A is virtual [M,768] gathered from x[M,256] rows n-1,n,n+1.
// Block: 256 threads (8 waves), tile 128(M) x 64(N), BK=32.
// B tile staged with GLOBAL_LOAD_ASYNC_TO_LDS_B128 (ASYNCcnt).
// ============================================================
template <int CONV>
__global__ __launch_bounds__(256) void k_gemm_bf16(
    const unsigned short* __restrict__ A,
    const unsigned short* __restrict__ Wt,
    const float* __restrict__ bias,
    float* __restrict__ C,
    int M, int K, int Nout)
{
    __shared__ alignas(16) unsigned short lA[128 * 32];
    __shared__ alignas(16) unsigned short lB[64 * 32];

    const int tid  = threadIdx.x;
    const int lane = tid & 31;
    const int wave = tid >> 5;
    const int wr = wave & 3;        // 4 wave-rows of 32
    const int wc = wave >> 2;       // 2 wave-cols of 32
    const int m0blk = blockIdx.x * 128;
    const int n0blk = blockIdx.y * 64;

    // per-thread staging coordinates
    const int arow  = tid >> 1;          // A: 2 threads per row
    const int ahalf = tid & 1;           // 16 bf16 each
    const int bn    = tid >> 2;          // B: 4 threads per n-row
    const int bkoff = (tid & 3) * 8;     // 8 bf16 each (16 bytes)
    const unsigned ldsB = (unsigned)(uintptr_t)(&lB[bn * 32 + bkoff]);

    v8f c00 = {}, c01 = {}, c10 = {}, c11 = {};

    for (int k0 = 0; k0 < K; k0 += 32) {
        // ---- B tile (weights): async global -> LDS, per-lane b128 (GVS mode)
        {
            unsigned voff = (unsigned)((((size_t)(n0blk + bn)) * K + k0 + bkoff) * 2u);
            asm volatile("global_load_async_to_lds_b128 %0, %1, %2"
                         :: "v"(ldsB), "v"(voff), "s"(Wt)
                         : "memory");
        }
        // ---- A tile: synchronous (needs zero-fill at edges for conv gather)
        {
            int gr = m0blk + arow;
            int kk = k0 + ahalf * 16;
            uint4 v0 = make_uint4(0, 0, 0, 0), v1 = v0;
            if (CONV == 0) {
                if (gr < M) {
                    const uint4* p = (const uint4*)(A + (size_t)gr * K + kk);
                    v0 = p[0]; v1 = p[1];
                }
            } else {
                int srow = gr + (kk >> 8) - 1;      // conv gather: rows n-1,n,n+1
                int colc = kk & 255;
                if (gr < M && srow >= 0 && srow < M) {
                    const uint4* p = (const uint4*)(A + (size_t)srow * 256 + colc);
                    v0 = p[0]; v1 = p[1];
                }
            }
            *(uint4*)&lA[arow * 32 + ahalf * 16]     = v0;
            *(uint4*)&lA[arow * 32 + ahalf * 16 + 8] = v1;
        }
        // prefetch next A k-slab
        if (CONV == 0 && k0 + 32 < K) {
            int gr = m0blk + arow;
            if (gr < M) __builtin_prefetch(A + (size_t)gr * K + k0 + 32, 0, 0);
        }
        // wait own async B transfers, then publish LDS to the workgroup
        asm volatile("s_wait_asynccnt 0x0" ::: "memory");
        __syncthreads();

        // ---- fragments (ISA 7.12.2 16-bit A layout / transposed-B layout)
        int ml = lane & 15, g = lane >> 4;
        union { v16bf v; uint4 u[2]; } a0, a1, b0, b1;
        int mrow0 = wr * 32, ncol0 = wc * 32;
        a0.u[0] = *(uint4*)&lA[(mrow0 + ml) * 32 + g * 8];
        a0.u[1] = *(uint4*)&lA[(mrow0 + ml) * 32 + 16 + g * 8];
        a1.u[0] = *(uint4*)&lA[(mrow0 + 16 + ml) * 32 + g * 8];
        a1.u[1] = *(uint4*)&lA[(mrow0 + 16 + ml) * 32 + 16 + g * 8];
        b0.u[0] = *(uint4*)&lB[(ncol0 + ml) * 32 + g * 8];
        b0.u[1] = *(uint4*)&lB[(ncol0 + ml) * 32 + 16 + g * 8];
        b1.u[0] = *(uint4*)&lB[(ncol0 + 16 + ml) * 32 + g * 8];
        b1.u[1] = *(uint4*)&lB[(ncol0 + 16 + ml) * 32 + 16 + g * 8];

        c00 = __builtin_amdgcn_wmma_f32_16x16x32_bf16(false, a0.v, false, b0.v, (short)0, c00, false, false);
        c01 = __builtin_amdgcn_wmma_f32_16x16x32_bf16(false, a0.v, false, b1.v, (short)0, c01, false, false);
        c10 = __builtin_amdgcn_wmma_f32_16x16x32_bf16(false, a1.v, false, b0.v, (short)0, c10, false, false);
        c11 = __builtin_amdgcn_wmma_f32_16x16x32_bf16(false, a1.v, false, b1.v, (short)0, c11, false, false);
        __syncthreads();
    }

    // ---- writeback: D layout (VGPR r -> M = r + 8*(lane/16); N = lane%16)
    int ml = lane & 15, half = lane >> 4;
    int mrow0 = wr * 32, ncol0 = wc * 32;
    auto storeTile = [&](v8f acc, int mbase, int nbase) {
        int gcol = n0blk + nbase + ml;
        float bv = bias[gcol];
        #pragma unroll
        for (int r = 0; r < 8; ++r) {
            int grow = m0blk + mbase + half * 8 + r;
            if (grow < M) C[(size_t)grow * Nout + gcol] = acc[r] + bv;
        }
    };
    storeTile(c00, mrow0,      ncol0);
    storeTile(c01, mrow0,      ncol0 + 16);
    storeTile(c10, mrow0 + 16, ncol0);
    storeTile(c11, mrow0 + 16, ncol0 + 16);
}

// ============================================================
// post-conv fuse: x = x0 + LN(gelu(xc));  writes xA f32 + bf16
// ============================================================
__global__ __launch_bounds__(256) void k_postconv(
    const float* __restrict__ xc, float* __restrict__ xA,
    unsigned short* __restrict__ xbf,
    const float* __restrict__ g, const float* __restrict__ b)
{
    __shared__ float s1[256], s2[256];
    int n = blockIdx.x, d = threadIdx.x;
    size_t i = (size_t)n * 256 + d;
    float ge = gelu_exact(xc[i]);
    s1[d] = ge; s2[d] = ge * ge;
    __syncthreads();
    for (int off = 128; off > 0; off >>= 1) {
        if (d < off) { s1[d] += s1[d + off]; s2[d] += s2[d + off]; }
        __syncthreads();
    }
    float mu  = s1[0] * (1.0f / 256.0f);
    float var = s2[0] * (1.0f / 256.0f) - mu * mu;
    float ln  = (ge - mu) * rsqrtf(var + 1e-5f) * g[d] + b[d];
    float v   = xA[i] + ln;
    xA[i] = v;
    xbf[i] = f2bf(v);
}

// ============================================================
// attention init: out rows = skip part of qkvs; zero segment max/sum
// ============================================================
__global__ __launch_bounds__(256) void k_attn_init(
    const float* __restrict__ qkvs, float* __restrict__ outb,
    unsigned* __restrict__ nmax, float* __restrict__ nsum, int n)
{
    int i = blockIdx.x * 256 + threadIdx.x;
    if (i < n * 256)
        outb[i] = qkvs[(size_t)(i >> 8) * 1024 + 768 + (i & 255)];
    if (i < n * 2) { nmax[i] = 0u; nsum[i] = 0.0f; }
}

// ============================================================
// edge logits: one wave per (edge, head); atomic segment max
// ============================================================
__global__ __launch_bounds__(256) void k_edge_logits(
    const float* __restrict__ qkvs, const int* __restrict__ eidx,
    const float* __restrict__ eattr, const float* __restrict__ ew,
    float* __restrict__ alpha, unsigned* __restrict__ nmax, int E)
{
    int wid = threadIdx.x >> 5, lane = threadIdx.x & 31;
    long long p = (long long)blockIdx.x * 8 + wid;
    if (p >= (long long)E * 2) return;
    int e = (int)(p >> 1), h = (int)(p & 1);
    int src = eidx[e], dst = eidx[E + e];
    float ea0 = eattr[e * 4], ea1 = eattr[e * 4 + 1];
    float ea2 = eattr[e * 4 + 2], ea3 = eattr[e * 4 + 3];
    const float* qrow = qkvs + (size_t)dst * 1024;
    const float* krow = qkvs + (size_t)src * 1024 + 256;
    float acc = 0.0f;
    #pragma unroll
    for (int j = 0; j < 4; ++j) {
        int col = h * 128 + lane + j * 32;
        float ev = ea0 * ew[col] + ea1 * ew[256 + col] + ea2 * ew[512 + col] + ea3 * ew[768 + col];
        acc += qrow[col] * (krow[col] + ev);
    }
    #pragma unroll
    for (int off = 16; off > 0; off >>= 1) acc += __shfl_xor(acc, off, 32);
    if (lane == 0) {
        float s = acc * 0.08838834764831845f;   // 1/sqrt(128)
        alpha[p] = s;
        atomicMax(&nmax[dst * 2 + h], encf(s));
    }
}

// ============================================================
// segment softmax numerator + atomic segment sum
// ============================================================
__global__ __launch_bounds__(256) void k_softmax_exp(
    const int* __restrict__ eidx, float* __restrict__ alpha,
    const unsigned* __restrict__ nmax, float* __restrict__ nsum, int E)
{
    int p = blockIdx.x * 256 + threadIdx.x;
    if (p >= E * 2) return;
    int e = p >> 1, h = p & 1;
    int dst = eidx[E + e];
    float ex = expf(alpha[p] - decf(nmax[dst * 2 + h]));
    alpha[p] = ex;
    atomicAdd(&nsum[dst * 2 + h], ex);
}

// ============================================================
// message scatter: out[dst] += (v[src]+e) * alpha / sum (atomic f32)
// ============================================================
__global__ __launch_bounds__(256) void k_scatter(
    const float* __restrict__ qkvs, const int* __restrict__ eidx,
    const float* __restrict__ eattr, const float* __restrict__ ew,
    const float* __restrict__ alpha, const float* __restrict__ nsum,
    float* __restrict__ outb, int E)
{
    int wid = threadIdx.x >> 5, lane = threadIdx.x & 31;
    long long p = (long long)blockIdx.x * 8 + wid;
    if (p >= (long long)E * 2) return;
    int e = (int)(p >> 1), h = (int)(p & 1);
    int src = eidx[e], dst = eidx[E + e];
    float coef = alpha[p] / (nsum[dst * 2 + h] + 1e-16f);
    float ea0 = eattr[e * 4], ea1 = eattr[e * 4 + 1];
    float ea2 = eattr[e * 4 + 2], ea3 = eattr[e * 4 + 3];
    const float* vrow = qkvs + (size_t)src * 1024 + 512;
    #pragma unroll
    for (int j = 0; j < 4; ++j) {
        int col = h * 128 + lane + j * 32;
        float ev = ea0 * ew[col] + ea1 * ew[256 + col] + ea2 * ew[512 + col] + ea3 * ew[768 + col];
        atomicAdd(&outb[(size_t)dst * 256 + col], (vrow[col] + ev) * coef);
    }
}

// ============================================================
// layer tail: x = gelu(LN(out + x_in)); writes xA f32 + bf16 (+ d_out)
// ============================================================
__global__ __launch_bounds__(256) void k_layer_final(
    const float* __restrict__ outb, float* __restrict__ xA,
    unsigned short* __restrict__ xbf,
    const float* __restrict__ g, const float* __restrict__ b,
    float* __restrict__ dout)
{
    __shared__ float s1[256], s2[256];
    int n = blockIdx.x, d = threadIdx.x;
    size_t i = (size_t)n * 256 + d;
    float t = outb[i] + xA[i];
    s1[d] = t; s2[d] = t * t;
    __syncthreads();
    for (int off = 128; off > 0; off >>= 1) {
        if (d < off) { s1[d] += s1[d + off]; s2[d] += s2[d + off]; }
        __syncthreads();
    }
    float mu  = s1[0] * (1.0f / 256.0f);
    float var = s2[0] * (1.0f / 256.0f) - mu * mu;
    float ln  = (t - mu) * rsqrtf(var + 1e-5f) * g[d] + b[d];
    float v   = gelu_exact(ln);
    xA[i] = v;
    xbf[i] = f2bf(v);
    if (dout) dout[i] = v;
}

// ============================================================
// launcher
// ============================================================
extern "C" void kernel_launch(void* const* d_in, const int* in_sizes, int n_in,
                              void* d_out, int out_size, void* d_ws, size_t ws_size,
                              hipStream_t stream)
{
    (void)in_sizes; (void)n_in; (void)out_size; (void)ws_size;
    const float* mx  = (const float*)d_in[0];
    const float* ms  = (const float*)d_in[1];
    const int*   eix = (const int*)d_in[2];
    const float* eat = (const float*)d_in[3];
    const float* bpw = (const float*)d_in[4];
    const float* bpb = (const float*)d_in[5];
    const float* psc = (const float*)d_in[6];
    const float* cw  = (const float*)d_in[7];
    const float* cb  = (const float*)d_in[8];
    const float* cg  = (const float*)d_in[9];
    const float* cbeta = (const float*)d_in[10];

    char* ws = (char*)d_ws;
    const size_t szX   = (size_t)NN * 256 * 4;     // 51.2 MB
    const size_t szXbf = (size_t)NN * 256 * 2;     // 25.6 MB
    const size_t szQ   = (size_t)NN * 1024 * 4;    // 204.8 MB
    float*          xA    = (float*)(ws);
    float*          xB    = (float*)(ws + szX);
    unsigned short* xbf   = (unsigned short*)(ws + 2 * szX);
    float*          qkvs  = (float*)(ws + 2 * szX + szXbf);
    unsigned short* Wt    = (unsigned short*)(ws + 2 * szX + szXbf + szQ);
    float*          biasb = (float*)(ws + 2 * szX + szXbf + szQ + 524288);
    float*          alpha = (float*)((char*)biasb + 4096);
    unsigned*       nmax  = (unsigned*)((char*)alpha + (size_t)EE * 2 * 4);
    float*          nsum  = (float*)((char*)nmax + (size_t)NN * 2 * 4);

    const int MT = (NN + 127) / 128;   // 391 M-tiles

    // phase 0: base proj + positional encoding
    k_base_proj<<<NN, 256, 0, stream>>>(mx, ms, bpw, bpb, psc, xA, xbf);

    // conv1d as GEMM [N,768]x[768,256]
    k_pack_conv<<<(256 * 768 + 255) / 256, 256, 0, stream>>>(cw, cb, Wt, biasb);
    k_gemm_bf16<1><<<dim3(MT, 4), 256, 0, stream>>>(xbf, Wt, biasb, xB, NN, 768, 256);
    k_postconv<<<NN, 256, 0, stream>>>(xB, xA, xbf, cg, cbeta);

    for (int layer = 0; layer < 2; ++layer) {
        const int base = 11 + layer * 11;
        const float* qw = (const float*)d_in[base + 0];
        const float* qb = (const float*)d_in[base + 1];
        const float* kw = (const float*)d_in[base + 2];
        const float* kb = (const float*)d_in[base + 3];
        const float* vw = (const float*)d_in[base + 4];
        const float* vb = (const float*)d_in[base + 5];
        const float* ew = (const float*)d_in[base + 6];
        const float* sw = (const float*)d_in[base + 7];
        const float* sb = (const float*)d_in[base + 8];
        const float* lg = (const float*)d_in[base + 9];
        const float* lb = (const float*)d_in[base + 10];

        // fused Q|K|V|skip GEMM [N,256]x[256,1024]
        k_pack_qkvs<<<(1024 * 256 + 255) / 256, 256, 0, stream>>>(
            qw, qb, kw, kb, vw, vb, sw, sb, Wt, biasb);
        k_gemm_bf16<0><<<dim3(MT, 16), 256, 0, stream>>>(xbf, Wt, biasb, qkvs, NN, 256, 1024);

        // graph attention
        k_attn_init<<<NN, 256, 0, stream>>>(qkvs, xB, nmax, nsum, NN);
        k_edge_logits<<<(EE * 2 + 7) / 8, 256, 0, stream>>>(qkvs, eix, eat, ew, alpha, nmax, EE);
        k_softmax_exp<<<(EE * 2 + 255) / 256, 256, 0, stream>>>(eix, alpha, nmax, nsum, EE);
        k_scatter<<<(EE * 2 + 7) / 8, 256, 0, stream>>>(qkvs, eix, eat, ew, alpha, nsum, xB, EE);

        // residual + LN + gelu
        k_layer_final<<<NN, 256, 0, stream>>>(xB, xA, xbf, lg, lb,
                                              (layer == 1) ? (float*)d_out : nullptr);
    }
}
